// WeightedHGTConv_50723563766182
// MI455X (gfx1250) — matrix operand order repaired and checked by reading, not compiled, not run on hardware
//
#include <hip/hip_runtime.h>
#include <hip/hip_bf16.h>

typedef __attribute__((ext_vector_type(16))) __bf16 v16bf;
typedef __attribute__((ext_vector_type(8)))  __bf16 v8bf;
typedef __attribute__((ext_vector_type(8)))  float  v8f;

// ---------------------------------------------------------------------------
// Kernel 0: init accumulators (d_out = 0, segment-max = -inf, Z = 0)
// ---------------------------------------------------------------------------
__global__ void k_init(float* __restrict__ outacc, unsigned* __restrict__ mbits,
                       float* __restrict__ Z, int N) {
    int gid = blockIdx.x * 256 + threadIdx.x;
    if (gid < N * 128) outacc[gid] = 0.0f;
    if (gid < N * 8) { mbits[gid] = 0xFF800000u; Z[gid] = 0.0f; }
}

// ---------------------------------------------------------------------------
// Kernel 1: node_inp f32 -> bf16 (row major, feeds WMMA A fragments directly)
// ---------------------------------------------------------------------------
__global__ void k_prep_a(const float* __restrict__ x, __bf16* __restrict__ a, int n) {
    int gid = blockIdx.x * 256 + threadIdx.x;
    if (gid < n) a[gid] = (__bf16)x[gid];
}

// ---------------------------------------------------------------------------
// Kernel 2: swizzle Wq/Wk/Wv (T,128,128 f32) into bf16 B-fragment layout:
//   [proj(3)][t(4)][ob(8)][kb(4)][lane(32)][i(16)]
//   lane l: n = l&15, kh = l>>4 ; element i -> k = kb*32 + kh*16 + i, o = ob*16 + n
// ---------------------------------------------------------------------------
__global__ void k_prep_w(const float* __restrict__ Wq, const float* __restrict__ Wk,
                         const float* __restrict__ Wv, __bf16* __restrict__ wb) {
    int gid = blockIdx.x * 256 + threadIdx.x;
    if (gid >= 3 * 4 * 8 * 4 * 32 * 16) return;
    int i    = gid & 15;
    int lane = (gid >> 4) & 31;
    int kb   = (gid >> 9) & 3;
    int ob   = (gid >> 11) & 7;
    int t    = (gid >> 14) & 3;
    int proj = gid >> 16;
    int k = kb * 32 + (lane >> 4) * 16 + i;
    int o = ob * 16 + (lane & 15);
    const float* W = (proj == 0) ? Wq : ((proj == 1) ? Wk : Wv);
    wb[gid] = (__bf16)W[((size_t)t * 128 + k) * 128 + o];
}

// ---------------------------------------------------------------------------
// Kernel 3: fused Q/K/V projection via bf16 WMMA. One 16-node tile per block,
// 4 waves split the 3 proj x 8 out-tiles. 4 weight slabs accumulated, selected
// per node by node_type in registers before the f32 store.
// ---------------------------------------------------------------------------
union AFrag { v16bf v; v8bf h[2]; };

__global__ __launch_bounds__(128) void k_qkv(
        const __bf16* __restrict__ Abf, const __bf16* __restrict__ Wbf,
        const int* __restrict__ node_type,
        const float* __restrict__ bq, const float* __restrict__ bk, const float* __restrict__ bv,
        float* __restrict__ Qf, float* __restrict__ Kf, float* __restrict__ Vf, int N) {
    int lane = threadIdx.x & 31;
    int wave = threadIdx.x >> 5;
    int tile = blockIdx.x;
    int m  = lane & 15;
    int kh = lane >> 4;

    int row = tile * 16 + m;
    if (row >= N) row = N - 1;          // uniform tail clamp (N%16==0 normally)

    // A fragments for the 4 k-blocks: two contiguous 16B bf16 loads each.
    v16bf a[4];
    const __bf16* arow = Abf + (size_t)row * 128;
#pragma unroll
    for (int kb = 0; kb < 4; ++kb) {
        const __bf16* p = arow + kb * 32 + kh * 16;
        AFrag af;
        af.h[0] = *(const v8bf*)p;
        af.h[1] = *(const v8bf*)(p + 8);
        a[kb] = af.v;
    }

    // node types for the 8 C-rows this lane owns (M = kh*8 + r)
    int trow[8];
#pragma unroll
    for (int r = 0; r < 8; ++r) {
        int node = tile * 16 + kh * 8 + r;
        trow[r] = (node < N) ? node_type[node] : 0;
    }

    for (int j = wave; j < 24; j += 4) {
        int proj = j >> 3;
        int ob   = j & 7;
        v8f acc0 = {}, acc1 = {}, acc2 = {}, acc3 = {};
#pragma unroll
        for (int kb = 0; kb < 4; ++kb) {
            const __bf16* wb = Wbf + (size_t)((((proj * 4 + 0) * 8 + ob) * 4 + kb)) * 512 + lane * 16;
            v16bf b0 = *(const v16bf*)(wb);
            v16bf b1 = *(const v16bf*)(wb + 8 * 512);   // flat t*8+ob slab ob+2
            v16bf b2 = *(const v16bf*)(wb + 16 * 512);  // flat t*8+ob slab ob+4
            v16bf b3 = *(const v16bf*)(wb + 24 * 512);  // flat t*8+ob slab ob+6
            acc0 = __builtin_amdgcn_wmma_f32_16x16x32_bf16(false, a[kb], false, b0, (short)0, acc0, false, false);
            acc1 = __builtin_amdgcn_wmma_f32_16x16x32_bf16(false, a[kb], false, b1, (short)0, acc1, false, false);
            acc2 = __builtin_amdgcn_wmma_f32_16x16x32_bf16(false, a[kb], false, b2, (short)0, acc2, false, false);
            acc3 = __builtin_amdgcn_wmma_f32_16x16x32_bf16(false, a[kb], false, b3, (short)0, acc3, false, false);
        }
        const float* bias = (proj == 0) ? bq : ((proj == 1) ? bk : bv);
        float* dstbuf     = (proj == 0) ? Qf : ((proj == 1) ? Kf : Vf);
        int n = lane & 15;
        int o = ob * 16 + n;
#pragma unroll
        for (int r = 0; r < 8; ++r) {
            int node = tile * 16 + kh * 8 + r;
            if (node < N) {
                int t = trow[r];
                float v = (t == 0) ? acc0[r] : ((t == 1) ? acc1[r] : ((t == 2) ? acc2[r] : acc3[r]));
                dstbuf[(size_t)node * 128 + o] = v + bias[t * 128 + o];
            }
        }
    }
}

// ---------------------------------------------------------------------------
// sign index helper:  es = (s<-1 || s==0) ? -2 : clip(s,-1,1)
//                     idx = es==-1 ? 0 : es==1 ? 1 : 2
// ---------------------------------------------------------------------------
__device__ __forceinline__ int sign_index(int s) {
    return ((s < -1) || (s == 0)) ? 2 : ((s < 0) ? 0 : 1);
}

// ---------------------------------------------------------------------------
// Kernel 4: per-(edge,head) attention score + segment max (atomic, sign-split)
// ---------------------------------------------------------------------------
__global__ void k_scores(const float* __restrict__ Qf, const float* __restrict__ Kf,
                         const int* __restrict__ ei, const int* __restrict__ etype,
                         const int* __restrict__ esign, const float* __restrict__ edist,
                         const float* __restrict__ rel_q, const float* __restrict__ rel_k,
                         const float* __restrict__ sign_k_neutral,
                         const float* __restrict__ rel_bias,
                         const float* __restrict__ p_alpha, const float* __restrict__ p_tau,
                         float* __restrict__ scores, unsigned* __restrict__ mbits, int E) {
    int gid = blockIdx.x * 256 + threadIdx.x;
    int e = gid >> 3, h = gid & 7;
    if (e >= E) return;
    int src = ei[e], dst = ei[E + e];
    int et  = etype[e];
    int sidx = sign_index(esign[e]);

    const float* qrow = Qf + (size_t)dst * 128 + h * 16;
    const float* krow = Kf + (size_t)src * 128 + h * 16;
    const float* rq   = rel_q + et * 128 + h * 16;
    const float* rk   = rel_k + et * 128 + h * 16;
    const float* sk   = sign_k_neutral + h * 16;

    float dot = 0.0f;
#pragma unroll
    for (int d = 0; d < 16; ++d) {
        float sgn = (sidx == 0) ? -1.0f : ((sidx == 1) ? 1.0f : sk[d]);
        dot += (qrow[d] * rq[d]) * (krow[d] * rk[d] * sgn);
    }
    float tau = p_tau[0] + 1e-9f;
    float phi = p_alpha[0] * expf(-edist[e] / tau);
    float sc  = dot * 0.25f + rel_bias[et * 8 + h] + phi;   // 1/sqrt(16) = 0.25

    scores[(size_t)e * 8 + h] = sc;
    unsigned* addr = mbits + (size_t)dst * 8 + h;
    int bits = __float_as_int(sc);
    if (bits >= 0) atomicMax((int*)addr, bits);             // non-negative floats
    else           atomicMin(addr, (unsigned)bits);         // negative floats
}

// ---------------------------------------------------------------------------
// Kernel 5: ex = exp(score - max[dst]); Z[dst] += ex; scores overwritten by ex
// ---------------------------------------------------------------------------
__global__ void k_softmax_num(const int* __restrict__ ei, const float* __restrict__ mvals,
                              float* __restrict__ scores, float* __restrict__ Z, int E) {
    int gid = blockIdx.x * 256 + threadIdx.x;
    int e = gid >> 3, h = gid & 7;
    if (e >= E) return;
    int dst = ei[E + e];
    float ex = expf(scores[gid] - mvals[(size_t)dst * 8 + h]);
    scores[gid] = ex;
    atomicAdd(&Z[(size_t)dst * 8 + h], ex);
}

// ---------------------------------------------------------------------------
// Kernel 6: weighted message scatter-add into d_out (one wave == one edge,
// each lane 4 contiguous output channels)
// ---------------------------------------------------------------------------
__global__ void k_message(const float* __restrict__ Vf, const int* __restrict__ ei,
                          const int* __restrict__ etype, const int* __restrict__ esign,
                          const float* __restrict__ rel_v, const float* __restrict__ sign_v_neutral,
                          const float* __restrict__ ex, const float* __restrict__ Z,
                          float* __restrict__ outacc, int E) {
    int gid = blockIdx.x * 256 + threadIdx.x;
    int e = gid >> 5;
    if (e >= E) return;
    int q = gid & 31;
    int o = q * 4;
    int h = o >> 4;
    int src = ei[e], dst = ei[E + e];
    int et  = etype[e];
    int sidx = sign_index(esign[e]);

    float attn = ex[(size_t)e * 8 + h] / (Z[(size_t)dst * 8 + h] + 1e-9f);
    float4 v  = *(const float4*)(Vf + (size_t)src * 128 + o);
    float4 rv = *(const float4*)(rel_v + et * 128 + o);
    float4 sv;
    if (sidx == 2) sv = *(const float4*)(sign_v_neutral + o);
    else { float f = (sidx == 0) ? -1.0f : 1.0f; sv = make_float4(f, f, f, f); }

    float* dp = outacc + (size_t)dst * 128 + o;
    atomicAdd(dp + 0, v.x * rv.x * sv.x * attn);
    atomicAdd(dp + 1, v.y * rv.y * sv.y * attn);
    atomicAdd(dp + 2, v.z * rv.z * sv.z * attn);
    atomicAdd(dp + 3, v.w * rv.w * sv.w * attn);
}

// ---------------------------------------------------------------------------
// Kernel 7: skip-mix + LayerNorm, in place on d_out. One wave per node.
// ---------------------------------------------------------------------------
__global__ __launch_bounds__(128) void k_finalize(
        const float* __restrict__ node_inp, const int* __restrict__ node_type,
        const float* __restrict__ skip, const float* __restrict__ gamma,
        const float* __restrict__ beta, float* __restrict__ out, int N) {
    int lane = threadIdx.x & 31;
    int wave = threadIdx.x >> 5;
    int n = blockIdx.x * 4 + wave;
    if (n >= N) return;
    int t = node_type[n];
    float alpha = 1.0f / (1.0f + expf(-skip[t]));
    float beta1 = 1.0f - alpha;

    float4 acc = *(const float4*)(out + (size_t)n * 128 + lane * 4);
    float4 inp = *(const float4*)(node_inp + (size_t)n * 128 + lane * 4);
    float4 mix;
    mix.x = alpha * acc.x + beta1 * inp.x;
    mix.y = alpha * acc.y + beta1 * inp.y;
    mix.z = alpha * acc.z + beta1 * inp.z;
    mix.w = alpha * acc.w + beta1 * inp.w;

    float sum = mix.x + mix.y + mix.z + mix.w;
    float sq  = mix.x * mix.x + mix.y * mix.y + mix.z * mix.z + mix.w * mix.w;
#pragma unroll
    for (int off = 16; off > 0; off >>= 1) {
        sum += __shfl_xor(sum, off, 32);
        sq  += __shfl_xor(sq, off, 32);
    }
    float mu   = sum * (1.0f / 128.0f);
    float var  = sq * (1.0f / 128.0f) - mu * mu;
    float rstd = rsqrtf(var + 1e-5f);

    float4 g = *(const float4*)(gamma + t * 128 + lane * 4);
    float4 b = *(const float4*)(beta + t * 128 + lane * 4);
    float4 r;
    r.x = (mix.x - mu) * rstd * g.x + b.x;
    r.y = (mix.y - mu) * rstd * g.y + b.y;
    r.z = (mix.z - mu) * rstd * g.z + b.z;
    r.w = (mix.w - mu) * rstd * g.w + b.w;
    *(float4*)(out + (size_t)n * 128 + lane * 4) = r;
}

// ---------------------------------------------------------------------------
extern "C" void kernel_launch(void* const* d_in, const int* in_sizes, int n_in,
                              void* d_out, int out_size, void* d_ws, size_t ws_size,
                              hipStream_t stream) {
    const float* node_inp  = (const float*)d_in[0];
    const int*   node_type = (const int*)d_in[1];
    const int*   ei        = (const int*)d_in[2];
    const int*   etype     = (const int*)d_in[3];
    const int*   esign     = (const int*)d_in[4];
    const float* edist     = (const float*)d_in[5];
    const float* Wq        = (const float*)d_in[6];
    const float* bq        = (const float*)d_in[7];
    const float* Wk        = (const float*)d_in[8];
    const float* bk        = (const float*)d_in[9];
    const float* Wv        = (const float*)d_in[10];
    const float* bv        = (const float*)d_in[11];
    const float* rel_q     = (const float*)d_in[12];
    const float* rel_k     = (const float*)d_in[13];
    const float* rel_v     = (const float*)d_in[14];
    const float* sk_neut   = (const float*)d_in[15];
    const float* sv_neut   = (const float*)d_in[16];
    const float* rel_bias  = (const float*)d_in[17];
    const float* p_alpha   = (const float*)d_in[18];
    const float* p_tau     = (const float*)d_in[19];
    const float* skip      = (const float*)d_in[20];
    const float* ln_gamma  = (const float*)d_in[21];
    const float* ln_beta   = (const float*)d_in[22];

    const int N = in_sizes[0] / 128;
    const int E = in_sizes[3];

    // workspace layout
    char* ws = (char*)d_ws;
    size_t szNodeF = (size_t)N * 128 * sizeof(float);
    float*    Qf     = (float*)(ws);                       ws += szNodeF;
    float*    Kf     = (float*)(ws);                       ws += szNodeF;
    float*    Vf     = (float*)(ws);                       ws += szNodeF;
    float*    scores = (float*)(ws);                       ws += (size_t)E * 8 * sizeof(float);
    unsigned* mbits  = (unsigned*)(ws);                    ws += (size_t)N * 8 * sizeof(float);
    float*    Z      = (float*)(ws);                       ws += (size_t)N * 8 * sizeof(float);
    __bf16*   Abf    = (__bf16*)(ws);                      ws += (size_t)N * 128 * sizeof(__bf16);
    __bf16*   Wbf    = (__bf16*)(ws);

    float* out = (float*)d_out;

    k_init<<<((size_t)N * 128 + 255) / 256, 256, 0, stream>>>(out, mbits, Z, N);
    k_prep_a<<<((size_t)N * 128 + 255) / 256, 256, 0, stream>>>(node_inp, Abf, N * 128);
    k_prep_w<<<(196608 + 255) / 256, 256, 0, stream>>>(Wq, Wk, Wv, Wbf);
    k_qkv<<<(N + 15) / 16, 128, 0, stream>>>(Abf, Wbf, node_type, bq, bk, bv, Qf, Kf, Vf, N);
    k_scores<<<((size_t)E * 8 + 255) / 256, 256, 0, stream>>>(
        Qf, Kf, ei, etype, esign, edist, rel_q, rel_k, sk_neut, rel_bias,
        p_alpha, p_tau, scores, mbits, E);
    k_softmax_num<<<((size_t)E * 8 + 255) / 256, 256, 0, stream>>>(
        ei, (const float*)mbits, scores, Z, E);
    k_message<<<((size_t)E * 32 + 255) / 256, 256, 0, stream>>>(
        Vf, ei, etype, esign, rel_v, sv_neut, scores, Z, out, E);
    k_finalize<<<(N + 3) / 4, 128, 0, stream>>>(
        node_inp, node_type, skip, ln_gamma, ln_beta, out, N);
}
